// STFSQ_78245714198640
// MI455X (gfx1250) — matrix operand order repaired and checked by reading, not compiled, not run on hardware
//
#include <hip/hip_runtime.h>
#include <math.h>

// ---------------------------------------------------------------------------
// Types for CDNA5 WMMA (wave32): bf16 16x16x32, f32 accumulate
// ---------------------------------------------------------------------------
typedef __bf16 bf16_t;
typedef bf16_t v16bf __attribute__((ext_vector_type(16)));
typedef float  v8f   __attribute__((ext_vector_type(8)));

#define DEVFN static __device__ __forceinline__

DEVFN float bf2f(unsigned short h) {
    union { unsigned int u; float f; } x;
    x.u = ((unsigned int)h) << 16;
    return x.f;
}
DEVFN unsigned short f2bf(float f) {
    union { float f; unsigned int u; } x;
    x.f = f;
    unsigned int u = x.u;
    unsigned int r = u + 0x7fffu + ((u >> 16) & 1u);   // round-to-nearest-even
    return (unsigned short)(r >> 16);
}

// ---------------------------------------------------------------------------
// B-row loader: 8 consecutive n values for one K row, branch-free boundary
// handling (unconditional loads + register zeroing; all activation buffers
// are preceded by the weight region in the workspace, so index -1 is mapped).
//   BMODE 0: B[k][n] = X[k][n]
//   BMODE 1: im2col k=3, segment length SEG
//   BMODE 2: im2col k=3 over 4x linear upsample of zhat[Cin][b*64+i], SEG=256
// ---------------------------------------------------------------------------
template<int BMODE, bool RELU_B, int SEG>
DEVFN void load_b_row(unsigned short* vals, const unsigned short* __restrict__ X,
                      int ldx, int kg, int K, int n0, int nb) {
    if (kg >= K) {
#pragma unroll
        for (int j = 0; j < 8; ++j) vals[j] = 0;
        return;
    }
    if (BMODE == 0) {
        uint4 raw = *(const uint4*)(X + (size_t)kg * ldx + n0 + nb);
        const unsigned short* pv = (const unsigned short*)&raw;
#pragma unroll
        for (int j = 0; j < 8; ++j) {
            unsigned short v = pv[j];
            if (RELU_B) v = (v & 0x8000u) ? (unsigned short)0 : v;
            vals[j] = v;
        }
    } else if (BMODE == 1) {
        const int ci = kg / 3;
        const int t  = kg - ci * 3;
        const unsigned short* row = X + (size_t)ci * ldx;
        const int base = n0 + nb + t - 1;                   // global col of j=0
        const int p0   = ((n0 + nb) & (SEG - 1)) + t - 1;   // seg pos of j=0
#pragma unroll
        for (int j = 0; j < 8; ++j) {
            unsigned short v = row[base + j];               // unconditional
            if (RELU_B) v = (v & 0x8000u) ? (unsigned short)0 : v;
            const int q = p0 + j;
            if (q < 0 || q >= SEG) v = 0;                   // v_cndmask
            vals[j] = v;
        }
    } else {  // BMODE 2
        const int ci = kg / 3;
        const int t  = kg - ci * 3;
        const unsigned short* row = X + (size_t)ci * ldx;
#pragma unroll
        for (int j = 0; j < 8; ++j) {
            const int nn = n0 + nb + j;
            const int b  = nn >> 8;
            const int l  = (nn & 255) + t - 1;              // -1..256
            float src = 0.25f * (float)l - 0.375f;
            src = fminf(fmaxf(src, 0.0f), 63.0f);           // safe even l=-1/256
            int   i0 = (int)src;
            float w  = src - (float)i0;
            int   i1 = (i0 + 1 < 63) ? (i0 + 1) : 63;
            float v0 = bf2f(row[b * 64 + i0]);
            float v1 = bf2f(row[b * 64 + i1]);
            unsigned short v = f2bf(v0 + (v1 - v0) * w);
            if (l < 0 || l >= 256) v = 0;
            vals[j] = v;
        }
    }
}

// ---------------------------------------------------------------------------
// Fused GEMM:  Y = epilogue( A[M,K] @ B[K,N] + bias[M] )
//   A: bf16 row-major weights (k=3 convs: K = 3*Cin, k = ci*3 + t matches the
//      (cout, cin, 3) flat layout of the reference weights)
//   X: bf16 activations, channel-major [Cin][ldx]
//   RELU_B : relu on B inputs (pre-activation resnet convs)
//   RELU_O : relu on output;  HAS_RES: add bf16 residual Res[M][N]
//   OUT 0: bf16 row-major [M][N]; 1: f32 row-major [M][N];
//       2: f32 in (B,C,L) layout (b = n>>8, l = n&255, C=512, L=256)
// Tiling: BM=128, BN=128, BK=32; 256 threads = 8 waves (4 m x 2 n), each wave
// computes 32x64 as 2x4 fragments of v_wmma_f32_16x16x32_bf16 (64 acc VGPRs).
// ---------------------------------------------------------------------------
template<int BMODE, bool RELU_B, bool RELU_O, bool HAS_RES, int OUT, int SEG>
__global__ __launch_bounds__(256)
void gemm_wmma(const unsigned short* __restrict__ A,
               const unsigned short* __restrict__ X,
               const float* __restrict__ bias,
               const unsigned short* __restrict__ Res,
               void* __restrict__ Y,
               int M, int K, int N, int ldx)
{
    __shared__ unsigned short As[128 * 32];   // [m][k]
    __shared__ unsigned short Bs[128 * 32];   // [n][k]

    const int tid   = threadIdx.x;
    const int tileM = blockIdx.y * 128;
    const int n0    = blockIdx.x * 128;

    const int wave = tid >> 5;
    const int wm   = wave >> 1;     // 0..3 -> m offset wm*32
    const int wn   = wave & 1;      // 0..1 -> n offset wn*64
    const int lane = tid & 31;
    const int hi   = lane >> 4;
    const int lr   = lane & 15;

    v8f acc[2][4] = {};

    // cooperative-load mappings
    const int arow = tid >> 1;             // 0..127
    const int acb  = (tid & 1) * 16;       // 0 or 16
    const int kh   = tid >> 4;             // 0..15 -> k rows {2kh, 2kh+1}
    const int bn   = (tid & 15) * 8;       // 0..120

    for (int kk = 0; kk < K; kk += 32) {
        // ---- stage A tile ----
        {
            const int gm = tileM + arow;
            unsigned short* dst = &As[arow * 32 + acb];
            if (gm < M && (kk + acb + 16) <= K) {
                const uint4* src = (const uint4*)(A + (size_t)gm * K + kk + acb);
                ((uint4*)dst)[0] = src[0];
                ((uint4*)dst)[1] = src[1];
            } else {
#pragma unroll
                for (int j = 0; j < 16; ++j) {
                    int c = kk + acb + j;
                    dst[j] = (gm < M && c < K) ? A[(size_t)gm * K + c]
                                               : (unsigned short)0;
                }
            }
        }
        // ---- stage B tile, transposed [n][k], k-pairs packed as b32 ----
        {
            unsigned short v0[8], v1[8];
            load_b_row<BMODE, RELU_B, SEG>(v0, X, ldx, kk + 2 * kh,     K, n0, bn);
            load_b_row<BMODE, RELU_B, SEG>(v1, X, ldx, kk + 2 * kh + 1, K, n0, bn);
#pragma unroll
            for (int j = 0; j < 8; ++j) {
                unsigned int packed =
                    (unsigned int)v0[j] | ((unsigned int)v1[j] << 16);
                *(unsigned int*)&Bs[(bn + j) * 32 + 2 * kh] = packed;
            }
        }
        __syncthreads();

        // ---- fragment loads + WMMA ----
        union FragU { v16bf v; uint4 q[2]; };
        FragU fa[2], fb[4];
#pragma unroll
        for (int fm = 0; fm < 2; ++fm) {
            const int r = wm * 32 + fm * 16 + lr;
            // 16-bit A layout: lane holds row M=lr; K = {hi*8+0..7, 16+hi*8+0..7}
            fa[fm].q[0] = *(const uint4*)&As[r * 32 + hi * 8];
            fa[fm].q[1] = *(const uint4*)&As[r * 32 + 16 + hi * 8];
        }
#pragma unroll
        for (int fn = 0; fn < 4; ++fn) {
            const int c = wn * 64 + fn * 16 + lr;
            // 16-bit B layout: lane holds col N=lr; K = {hi*8+0..7, 16+hi*8+0..7}
            fb[fn].q[0] = *(const uint4*)&Bs[c * 32 + hi * 8];
            fb[fn].q[1] = *(const uint4*)&Bs[c * 32 + 16 + hi * 8];
        }
#pragma unroll
        for (int fm = 0; fm < 2; ++fm)
#pragma unroll
            for (int fn = 0; fn < 4; ++fn)
                acc[fm][fn] = __builtin_amdgcn_wmma_f32_16x16x32_bf16(
                    false, fa[fm].v, false, fb[fn].v,
                    (short)0, acc[fm][fn], false, false);

        __syncthreads();
    }

    // ---- epilogue ----
#pragma unroll
    for (int fm = 0; fm < 2; ++fm) {
#pragma unroll
        for (int fn = 0; fn < 4; ++fn) {
            const int ng = n0 + wn * 64 + fn * 16 + lr;
#pragma unroll
            for (int v = 0; v < 8; ++v) {
                const int mg = tileM + wm * 32 + fm * 16 + hi * 8 + v;
                if (mg < M) {
                    float val = acc[fm][fn][v] + bias[mg];
                    if (RELU_O) val = fmaxf(val, 0.0f);
                    if (HAS_RES) val += bf2f(Res[(size_t)mg * N + ng]);
                    if (OUT == 0) {
                        ((unsigned short*)Y)[(size_t)mg * N + ng] = f2bf(val);
                    } else if (OUT == 1) {
                        ((float*)Y)[(size_t)mg * N + ng] = val;
                    } else {
                        const int b = ng >> 8, l = ng & 255;
                        ((float*)Y)[((size_t)b * 512 + mg) * 256 + l] = val;
                    }
                }
            }
        }
    }
}

// ---------------------------------------------------------------------------
// Auxiliary kernels
// ---------------------------------------------------------------------------
__global__ void k_cvt_bf16(const float* __restrict__ in,
                           unsigned short* __restrict__ out, int n) {
    for (int i = blockIdx.x * blockDim.x + threadIdx.x; i < n;
         i += gridDim.x * blockDim.x)
        out[i] = f2bf(in[i]);
}

__global__ void k_zero_f32(float* __restrict__ p, int n) {
    for (int i = blockIdx.x * blockDim.x + threadIdx.x; i < n;
         i += gridDim.x * blockDim.x)
        p[i] = 0.0f;
}

// f (B=256, C=512, L=256) -> avgpool4 -> bf16 channel-major [512][B*64]
__global__ void k_avgpool(const float* __restrict__ f,
                          unsigned short* __restrict__ out) {
    const int total = 512 * 16384;
    for (int i = blockIdx.x * blockDim.x + threadIdx.x; i < total;
         i += gridDim.x * blockDim.x) {
        const int c = i >> 14;
        const int n = i & 16383;
        const int b = n >> 6;
        const int p = n & 63;
        const float* s = f + ((size_t)b * 512 + c) * 256 + p * 4;
        out[i] = f2bf(0.25f * (s[0] + s[1] + s[2] + s[3]));
    }
}

// FSQ quantize: LEVELS=[5,5,5,5], half_l=1.998, offset=shift=0, half_width=2
__global__ void k_quantize(const float* __restrict__ Zf,
                           const float* __restrict__ mask,
                           unsigned short* __restrict__ C4,
                           float* __restrict__ cnt) {
    const int i = blockIdx.x * blockDim.x + threadIdx.x;
    if (i >= 16384) return;
    const int b = i >> 6, p = i & 63;
    const int basis[4] = {1, 5, 25, 125};
    int idx = 0;
#pragma unroll
    for (int c = 0; c < 4; ++c) {
        float z = Zf[c * 16384 + i];
        float bounded = tanhf(z) * 1.998f;
        float q = rintf(bounded);               // in {-2..2}
        C4[c * 16384 + i] = f2bf(q * 0.5f);     // exact in bf16
        idx += (int)(q + 2.0f) * basis[c];
    }
    const float valid = 1.0f - mask[b * 256 + p * 4];  // new_mask = m[:, :, p*4]
    atomicAdd(&cnt[idx], valid);
}

__global__ void k_sse(const float* __restrict__ h, const float* __restrict__ f,
                      float* __restrict__ sse, int n) {
    __shared__ float sh[256];
    float s = 0.0f;
    for (int i = blockIdx.x * blockDim.x + threadIdx.x; i < n;
         i += gridDim.x * blockDim.x) {
        const float d = h[i] - f[i];
        s += d * d;
    }
    sh[threadIdx.x] = s;
    __syncthreads();
    for (int o = 128; o > 0; o >>= 1) {
        if (threadIdx.x < o) sh[threadIdx.x] += sh[threadIdx.x + o];
        __syncthreads();
    }
    if (threadIdx.x == 0) atomicAdd(sse, sh[0]);
}

__global__ void k_final(const float* __restrict__ cnt,
                        const float* __restrict__ sse,
                        float* __restrict__ out_tail) {
    __shared__ float sh[1024];
    const int t = threadIdx.x;
    const float c = (t < 625) ? cnt[t] : 0.0f;
    sh[t] = c;
    __syncthreads();
    for (int o = 512; o > 0; o >>= 1) {
        if (t < o) sh[t] += sh[t + o];
        __syncthreads();
    }
    const float total = sh[0];
    __syncthreads();
    const float p = (t < 625) ? (c / total) : 0.0f;
    sh[t] = p * logf(p + 1e-7f);
    __syncthreads();
    for (int o = 512; o > 0; o >>= 1) {
        if (t < o) sh[t] += sh[t + o];
        __syncthreads();
    }
    if (t == 0) {
        out_tail[0] = 2.0f * sse[0] / 33554432.0f;  // mse1 + mse2
        out_tail[1] = expf(-sh[0]);                 // perplexity
    }
}

// ---------------------------------------------------------------------------
// Host launcher
// ---------------------------------------------------------------------------
extern "C" void kernel_launch(void* const* d_in, const int* in_sizes, int n_in,
                              void* d_out, int out_size, void* d_ws,
                              size_t ws_size, hipStream_t stream) {
    (void)in_sizes; (void)n_in; (void)out_size; (void)ws_size;

    // JAX pytree (sorted dict keys) input order
    const float* f    = (const float*)d_in[0];
    const float* mask = (const float*)d_in[1];
    const float* pi_b0    = (const float*)d_in[2];
    const float* pi_b1    = (const float*)d_in[3];
    const float* pi_b2    = (const float*)d_in[4];
    const float* pi_bb1_0 = (const float*)d_in[5];
    const float* pi_bb2_0 = (const float*)d_in[6];
    const float* pi_bw1_0 = (const float*)d_in[7];
    const float* pi_bw2_0 = (const float*)d_in[8];
    const float* pi_bb1_1 = (const float*)d_in[9];
    const float* pi_bb2_1 = (const float*)d_in[10];
    const float* pi_bw1_1 = (const float*)d_in[11];
    const float* pi_bw2_1 = (const float*)d_in[12];
    const float* pi_w0    = (const float*)d_in[13];
    const float* pi_w1    = (const float*)d_in[14];
    const float* pi_w2    = (const float*)d_in[15];
    const float* po_b0    = (const float*)d_in[16];
    const float* po_b1    = (const float*)d_in[17];
    const float* po_b2    = (const float*)d_in[18];
    const float* po_bb1_0 = (const float*)d_in[19];
    const float* po_bb2_0 = (const float*)d_in[20];
    const float* po_bw1_0 = (const float*)d_in[21];
    const float* po_bw2_0 = (const float*)d_in[22];
    const float* po_bb1_1 = (const float*)d_in[23];
    const float* po_bb2_1 = (const float*)d_in[24];
    const float* po_bw1_1 = (const float*)d_in[25];
    const float* po_bw2_1 = (const float*)d_in[26];
    const float* po_w0    = (const float*)d_in[27];
    const float* po_w1    = (const float*)d_in[28];
    const float* po_w2    = (const float*)d_in[29];
    const float* phi_b    = (const float*)d_in[30];
    const float* phi_w    = (const float*)d_in[31];

    char*  ws  = (char*)d_ws;
    size_t off = 0;
    auto alloc = [&](size_t bytes) -> void* {
        void* p = ws + off;
        off += (bytes + 255) & ~(size_t)255;
        return p;
    };
    auto cvt = [&](const float* src, int n) -> unsigned short* {
        unsigned short* dst = (unsigned short*)alloc((size_t)n * 2);
        k_cvt_bf16<<<dim3((n + 255) / 256), dim3(256), 0, stream>>>(src, dst, n);
        return dst;
    };

    const int Nt = 16384;          // B * P
    const int N2 = 65536;          // B * L

    // bf16 weights
    unsigned short* W_pi_w0    = cvt(pi_w0, 512 * 512);
    unsigned short* W_pi_bw1_0 = cvt(pi_bw1_0, 512 * 512 * 3);
    unsigned short* W_pi_bw2_0 = cvt(pi_bw2_0, 512 * 512);
    unsigned short* W_pi_bw1_1 = cvt(pi_bw1_1, 512 * 512 * 3);
    unsigned short* W_pi_bw2_1 = cvt(pi_bw2_1, 512 * 512);
    unsigned short* W_pi_w1    = cvt(pi_w1, 256 * 512);
    unsigned short* W_pi_w2    = cvt(pi_w2, 4 * 256);
    unsigned short* W_po_w0    = cvt(po_w0, 256 * 4);
    unsigned short* W_po_w1    = cvt(po_w1, 512 * 256);
    unsigned short* W_po_bw1_0 = cvt(po_bw1_0, 512 * 512 * 3);
    unsigned short* W_po_bw2_0 = cvt(po_bw2_0, 512 * 512);
    unsigned short* W_po_bw1_1 = cvt(po_bw1_1, 512 * 512 * 3);
    unsigned short* W_po_bw2_1 = cvt(po_bw2_1, 512 * 512);
    unsigned short* W_po_w2    = cvt(po_w2, 512 * 512);
    unsigned short* W_phi      = cvt(phi_w, 512 * 512 * 3);

    // activations / scratch (all after weights -> index -1 reads stay mapped)
    unsigned short* ACT0 = (unsigned short*)alloc((size_t)512 * Nt * 2);
    unsigned short* ACT1 = (unsigned short*)alloc((size_t)512 * Nt * 2);
    unsigned short* ACT2 = (unsigned short*)alloc((size_t)512 * Nt * 2);
    float*          Zf   = (float*)alloc((size_t)4 * Nt * 4);
    unsigned short* C4   = (unsigned short*)alloc((size_t)4 * Nt * 2);
    float*          cnt  = (float*)alloc(626 * 4);
    float*          sse  = cnt + 625;

    k_zero_f32<<<dim3(3), dim3(256), 0, stream>>>(cnt, 626);
    k_avgpool<<<dim3((512 * Nt + 255) / 256), dim3(256), 0, stream>>>(f, ACT0);

    const dim3 blk(256);
    const dim3 g512(Nt / 128, 4), g256(Nt / 128, 2), g4(Nt / 128, 1);

    // ---- project_in ----
    gemm_wmma<0, false, false, false, 0, 64><<<g512, blk, 0, stream>>>(
        W_pi_w0, ACT0, pi_b0, nullptr, ACT1, 512, 512, Nt, Nt);          // X0
    gemm_wmma<1, true, false, false, 0, 64><<<g512, blk, 0, stream>>>(
        W_pi_bw1_0, ACT1, pi_bb1_0, nullptr, ACT2, 512, 1536, Nt, Nt);   // conv3(relu X0)
    gemm_wmma<0, true, false, true, 0, 64><<<g512, blk, 0, stream>>>(
        W_pi_bw2_0, ACT2, pi_bb2_0, ACT1, ACT0, 512, 512, Nt, Nt);       // X1 = X0 + ...
    gemm_wmma<1, true, false, false, 0, 64><<<g512, blk, 0, stream>>>(
        W_pi_bw1_1, ACT0, pi_bb1_1, nullptr, ACT1, 512, 1536, Nt, Nt);
    gemm_wmma<0, true, false, true, 0, 64><<<g512, blk, 0, stream>>>(
        W_pi_bw2_1, ACT1, pi_bb2_1, ACT0, ACT2, 512, 512, Nt, Nt);       // X2
    gemm_wmma<0, false, true, false, 0, 64><<<g256, blk, 0, stream>>>(
        W_pi_w1, ACT2, pi_b1, nullptr, ACT0, 256, 512, Nt, Nt);          // X3 = relu(...)
    gemm_wmma<0, false, false, false, 1, 64><<<g4, blk, 0, stream>>>(
        W_pi_w2, ACT0, pi_b2, nullptr, Zf, 4, 256, Nt, Nt);              // z (f32)

    // ---- quantize + code counts ----
    k_quantize<<<dim3((Nt + 255) / 256), dim3(256), 0, stream>>>(Zf, mask, C4, cnt);

    // ---- project_out ----
    gemm_wmma<0, false, true, false, 0, 64><<<g256, blk, 0, stream>>>(
        W_po_w0, C4, po_b0, nullptr, ACT0, 256, 4, Nt, Nt);              // Y0 = relu(...)
    gemm_wmma<0, false, false, false, 0, 64><<<g512, blk, 0, stream>>>(
        W_po_w1, ACT0, po_b1, nullptr, ACT1, 512, 256, Nt, Nt);          // Y1
    gemm_wmma<1, true, false, false, 0, 64><<<g512, blk, 0, stream>>>(
        W_po_bw1_0, ACT1, po_bb1_0, nullptr, ACT2, 512, 1536, Nt, Nt);
    gemm_wmma<0, true, false, true, 0, 64><<<g512, blk, 0, stream>>>(
        W_po_bw2_0, ACT2, po_bb2_0, ACT1, ACT0, 512, 512, Nt, Nt);       // Y2
    gemm_wmma<1, true, false, false, 0, 64><<<g512, blk, 0, stream>>>(
        W_po_bw1_1, ACT0, po_bb1_1, nullptr, ACT1, 512, 1536, Nt, Nt);
    gemm_wmma<0, true, false, true, 0, 64><<<g512, blk, 0, stream>>>(
        W_po_bw2_1, ACT1, po_bb2_1, ACT0, ACT2, 512, 512, Nt, Nt);       // Y3
    gemm_wmma<0, false, false, false, 0, 64><<<g512, blk, 0, stream>>>(
        W_po_w2, ACT2, po_b2, nullptr, ACT0, 512, 512, Nt, Nt);          // zhat

    // ---- phi conv over upsampled signal, fused interp, f32 out -> d_out ----
    gemm_wmma<2, false, false, false, 2, 256><<<dim3(N2 / 128, 4), blk, 0, stream>>>(
        W_phi, ACT0, phi_b, nullptr, d_out, 512, 1536, N2, Nt);

    // ---- losses ----
    k_sse<<<dim3(4096), dim3(256), 0, stream>>>((const float*)d_out, f, sse,
                                                33554432);
    k_final<<<dim3(1), dim3(1024), 0, stream>>>(cnt, sse,
                                                ((float*)d_out) + 33554432);
}
